// GNN_ToxModel_54228257079879
// MI455X (gfx1250) — matrix operand order repaired and checked by reading, not compiled, not run on hardware
//
#include <hip/hip_runtime.h>
#include <hip/hip_bf16.h>

typedef __attribute__((ext_vector_type(16))) _Float16 v16h;
typedef __attribute__((ext_vector_type(8)))  float    v8f;

// ---------------- utility kernels ----------------

__global__ void zero_f32(float* __restrict__ p, int n) {
  int stride = gridDim.x * blockDim.x;
  for (int i = blockIdx.x * blockDim.x + threadIdx.x; i < n; i += stride)
    p[i] = 0.0f;
}

// agg[node*da + c] = (c < dx) ? x[node*dx + c] : 0   (zero-padded copy)
__global__ void init_agg(const float* __restrict__ x, float* __restrict__ agg,
                         int n, int dx, int da) {
  int total = n * da;
  int stride = gridDim.x * blockDim.x;
  for (int i = blockIdx.x * blockDim.x + threadIdx.x; i < total; i += stride) {
    int node = i / da;
    int c = i - node * da;
    agg[i] = (c < dx) ? x[node * dx + c] : 0.0f;
  }
}

// Pack weight W[K x N] (f32) into f16 WMMA B-fragment order, zero-padding to
// Kpad x Npad:  P[((tn*kB + kb)*32 + lane)*16 + j] = W[kb*32+8*(lane>>4)+(j&7)+16*(j>>3)][tn*16+(lane&15)]
__global__ void pack_B(const float* __restrict__ W, _Float16* __restrict__ P,
                       int K, int N, int Kpad, int Npad) {
  int kB = Kpad >> 5;
  int total = (Npad >> 4) * kB * 512;
  int stride = gridDim.x * blockDim.x;
  for (int i = blockIdx.x * blockDim.x + threadIdx.x; i < total; i += stride) {
    int j = i & 15;
    int lane = (i >> 4) & 31;
    int blk = i >> 9;
    int kb = blk % kB;
    int tn = blk / kB;
    int col = tn * 16 + (lane & 15);
    int kk = kb * 32 + (lane >> 4) * 8 + (j & 7) + ((j >> 3) << 4);
    P[i] = (kk < K && col < N) ? (_Float16)W[kk * N + col] : (_Float16)0;
  }
}

__global__ void pad_bias(const float* __restrict__ b, float* __restrict__ bp,
                         int n, int npad) {
  int i = blockIdx.x * blockDim.x + threadIdx.x;
  if (i < npad) bp[i] = (i < n) ? b[i] : 0.0f;
}

// dst[r*ncdst + c] = src[r*ncsrc + c]
__global__ void extract_cols(const float* __restrict__ src, float* __restrict__ dst,
                             int rows, int ncsrc, int ncdst) {
  int total = rows * ncdst;
  int stride = gridDim.x * blockDim.x;
  for (int i = blockIdx.x * blockDim.x + threadIdx.x; i < total; i += stride) {
    int r = i / ncdst;
    int c = i - r * ncdst;
    dst[i] = src[r * ncsrc + c];
  }
}

// ---------------- GIN aggregation: agg[dst] += x[src] ----------------

__global__ void scatter_add_edges(const float* __restrict__ x,
                                  const long long* __restrict__ src,
                                  const long long* __restrict__ dst,
                                  float* __restrict__ agg,
                                  int nE, int d4, int dx, int da) {
  long long total = (long long)nE * d4;
  long long stride = (long long)gridDim.x * blockDim.x;
  for (long long i = (long long)blockIdx.x * blockDim.x + threadIdx.x; i < total; i += stride) {
    int e = (int)(i / d4);
    int c = (int)(i % d4) * 4;
    int s = (int)src[e];
    int t = (int)dst[e];
    const float4 v = *(const float4*)(x + (size_t)s * dx + c);
    float* p = agg + (size_t)t * da + c;
    atomicAdd(p + 0, v.x);
    atomicAdd(p + 1, v.y);
    atomicAdd(p + 2, v.z);
    atomicAdd(p + 3, v.w);
  }
}

// ---------------- global add pool ----------------

__global__ void pool_add(const float* __restrict__ h,
                         const long long* __restrict__ batch,
                         float* __restrict__ ge, int n, int c4, int c) {
  long long total = (long long)n * c4;
  long long stride = (long long)gridDim.x * blockDim.x;
  for (long long i = (long long)blockIdx.x * blockDim.x + threadIdx.x; i < total; i += stride) {
    int node = (int)(i / c4);
    int ch = (int)(i % c4) * 4;
    int g = (int)batch[node];
    const float4 v = *(const float4*)(h + (size_t)node * c + ch);
    float* p = ge + (size_t)g * c + ch;
    atomicAdd(p + 0, v.x);
    atomicAdd(p + 1, v.y);
    atomicAdd(p + 2, v.z);
    atomicAdd(p + 3, v.w);
  }
}

// ---------------- WMMA GEMM (padded, guard-free inner loop) ----------------
// Requirements: M%16==0, N%16==0, K%32==0, (N/16)%NT==0.
// One wave computes a 16 x (16*NT) strip: one A fragment reused by NT WMMAs.

template <int NT>
__global__ void wmma_gemm_pack(const float* __restrict__ A,
                               const _Float16* __restrict__ Bp,
                               const float* __restrict__ bias,
                               float* __restrict__ C,
                               int M, int N, int K, int relu) {
  const int lane = threadIdx.x & 31;
  const int wave = threadIdx.x >> 5;
  const int nGroups = (N >> 4) / NT;
  const int kBlocks = K >> 5;
  const int tile = blockIdx.x * (blockDim.x >> 5) + wave;  // wave-uniform
  if (tile >= (M >> 4) * nGroups) return;                  // whole-wave exit

  const int tm = tile / nGroups;
  const int tg = tile % nGroups;
  const int half = lane >> 4;
  const int idx = lane & 15;
  const unsigned row = tm * 16 + idx;

  v8f acc[NT];
#pragma unroll
  for (int t = 0; t < NT; ++t) acc[t] = (v8f){0.f, 0.f, 0.f, 0.f, 0.f, 0.f, 0.f, 0.f};

  const float* arow = A + (size_t)row * K + half * 8;
  const _Float16* bbase = Bp + ((size_t)(tg * NT) * kBlocks * 32 + lane) * 16;

  for (int kb = 0; kb < kBlocks; ++kb) {
    __builtin_prefetch(arow + 32, 0, 0);  // speculative next K-slab
    const float4 a0 = *(const float4*)(arow + 0);
    const float4 a1 = *(const float4*)(arow + 4);
    const float4 a2 = *(const float4*)(arow + 16);
    const float4 a3 = *(const float4*)(arow + 20);
    arow += 32;
    v16h a;
    a[0] = (_Float16)a0.x;  a[1] = (_Float16)a0.y;
    a[2] = (_Float16)a0.z;  a[3] = (_Float16)a0.w;
    a[4] = (_Float16)a1.x;  a[5] = (_Float16)a1.y;
    a[6] = (_Float16)a1.z;  a[7] = (_Float16)a1.w;
    a[8] = (_Float16)a2.x;  a[9] = (_Float16)a2.y;
    a[10] = (_Float16)a2.z; a[11] = (_Float16)a2.w;
    a[12] = (_Float16)a3.x; a[13] = (_Float16)a3.y;
    a[14] = (_Float16)a3.z; a[15] = (_Float16)a3.w;
#pragma unroll
    for (int t = 0; t < NT; ++t) {
      const v16h b = *(const v16h*)(bbase + (size_t)t * kBlocks * 512);
      acc[t] = __builtin_amdgcn_wmma_f32_16x16x32_f16(
          false, a, false, b, (short)0, acc[t], false, false);
    }
    bbase += 512;  // next k-block fragment (32 lanes * 16 f16)
  }

#pragma unroll
  for (int t = 0; t < NT; ++t) {
    const int col = (tg * NT + t) * 16 + idx;
    const float bv = bias[col];
#pragma unroll
    for (int r = 0; r < 8; ++r) {
      const int m = tm * 16 + r + half * 8;
      float v = acc[t][r] + bv;
      if (relu) v = fmaxf(v, 0.0f);
      C[(size_t)m * N + col] = v;
    }
  }
}

// ---------------- BatchNorm (training-mode stats) ----------------

__global__ void bn_stats(const float* __restrict__ h, float* __restrict__ sums,
                         int n, int c) {
  int t = blockIdx.x * blockDim.x + threadIdx.x;
  int ch = t % c;
  int r0 = t / c;
  int rstride = (gridDim.x * blockDim.x) / c;
  float s = 0.0f, s2 = 0.0f;
  for (int r = r0; r < n; r += rstride) {
    float v = h[(size_t)r * c + ch];
    s += v;
    s2 += v * v;
  }
  atomicAdd(&sums[ch], s);
  atomicAdd(&sums[c + ch], s2);
}

__global__ void bn_apply_relu(float* __restrict__ h, const float* __restrict__ sums,
                              const float* __restrict__ gamma, const float* __restrict__ beta,
                              int n, int c) {
  int total = n * c;
  int stride = gridDim.x * blockDim.x;
  float inv_n = 1.0f / (float)n;
  for (int i = blockIdx.x * blockDim.x + threadIdx.x; i < total; i += stride) {
    int ch = i % c;
    float mean = sums[ch] * inv_n;
    float var = sums[c + ch] * inv_n - mean * mean;
    float v = (h[i] - mean) * rsqrtf(var + 1e-5f) * gamma[ch] + beta[ch];
    h[i] = fmaxf(v, 0.0f);
  }
}

// ---------------- host orchestration ----------------

extern "C" void kernel_launch(void* const* d_in, const int* in_sizes, int n_in,
                              void* d_out, int out_size, void* d_ws, size_t ws_size,
                              hipStream_t stream) {
  const float* x = (const float*)d_in[0];
  const long long* edge = (const long long*)d_in[1];
  const long long* batch = (const long long*)d_in[2];

  const int nNodes = in_sizes[2];               // 100000 (multiple of 16)
  const int nEdges = in_sizes[1] / 2;           // 1600000
  const int inDim = in_sizes[0] / nNodes;       // 12
  const int hid = in_sizes[4];                  // 128
  const int r1d = in_sizes[22];                 // 512
  const int r2d = in_sizes[24];                 // 256
  const int tasks = in_sizes[26];               // 12
  const int tasksPad = 16;
  const int G = out_size / tasks;               // 1024

  const long long* srcIdx = edge;
  const long long* dstIdx = edge + nEdges;

  const float* w1[3] = {(const float*)d_in[3], (const float*)d_in[9],  (const float*)d_in[15]};
  const float* b1[3] = {(const float*)d_in[4], (const float*)d_in[10], (const float*)d_in[16]};
  const float* w2[3] = {(const float*)d_in[5], (const float*)d_in[11], (const float*)d_in[17]};
  const float* b2[3] = {(const float*)d_in[6], (const float*)d_in[12], (const float*)d_in[18]};
  const float* gm[3] = {(const float*)d_in[7], (const float*)d_in[13], (const float*)d_in[19]};
  const float* bt[3] = {(const float*)d_in[8], (const float*)d_in[14], (const float*)d_in[20]};
  const float* rw1 = (const float*)d_in[21];
  const float* rb1 = (const float*)d_in[22];
  const float* rw2 = (const float*)d_in[23];
  const float* rb2 = (const float*)d_in[24];
  const float* rw3 = (const float*)d_in[25];
  const float* rb3 = (const float*)d_in[26];

  // carve workspace
  char* wsp = (char*)d_ws;
  size_t off = 0;
  auto carve = [&](size_t bytes) -> void* {
    void* p = wsp + off;
    off += (bytes + 255) & ~(size_t)255;
    return p;
  };
  float* X    = (float*)carve((size_t)nNodes * hid * 4);   // activations / GEMM2 out
  float* AGG  = (float*)carve((size_t)nNodes * hid * 4);   // aggregation (layers 1,2)
  float* MID  = (float*)carve((size_t)nNodes * hid * 4);   // GEMM1 out
  float* AGGP = (float*)carve((size_t)nNodes * 32 * 4);    // layer-0 agg, K padded 12->32
  float* GE   = (float*)carve((size_t)G * hid * 4);
  float* R1   = (float*)carve((size_t)G * r1d * 4);
  float* R2   = (float*)carve((size_t)G * r2d * 4);
  float* R3   = (float*)carve((size_t)G * tasksPad * 4);   // padded task head out
  float* SUMS = (float*)carve((size_t)2 * hid * 4);
  float* RB3P = (float*)carve((size_t)tasksPad * 4);
  _Float16* w1p[3], *w2p[3];
  w1p[0] = (_Float16*)carve((size_t)32 * hid * 2);         // Kpad=32
  w2p[0] = (_Float16*)carve((size_t)hid * hid * 2);
  for (int l = 1; l < 3; ++l) {
    w1p[l] = (_Float16*)carve((size_t)hid * hid * 2);
    w2p[l] = (_Float16*)carve((size_t)hid * hid * 2);
  }
  _Float16* rw1p = (_Float16*)carve((size_t)hid * r1d * 2);
  _Float16* rw2p = (_Float16*)carve((size_t)r1d * r2d * 2);
  _Float16* rw3p = (_Float16*)carve((size_t)r2d * tasksPad * 2);

  auto pack = [&](const float* W, _Float16* P, int K, int N, int Kpad, int Npad) {
    int total = (Npad / 16) * (Kpad / 32) * 512;
    pack_B<<<(total + 255) / 256, 256, 0, stream>>>(W, P, K, N, Kpad, Npad);
  };
  pack(w1[0], w1p[0], inDim, hid, 32, hid);
  pack(w2[0], w2p[0], hid, hid, hid, hid);
  for (int l = 1; l < 3; ++l) {
    pack(w1[l], w1p[l], hid, hid, hid, hid);
    pack(w2[l], w2p[l], hid, hid, hid, hid);
  }
  pack(rw1, rw1p, hid, r1d, hid, r1d);
  pack(rw2, rw2p, r1d, r2d, r1d, r2d);
  pack(rw3, rw3p, r2d, tasks, r2d, tasksPad);
  pad_bias<<<1, 64, 0, stream>>>(rb3, RB3P, tasks, tasksPad);

  auto gemm4 = [&](const float* A, const _Float16* Bp, const float* bias, float* C,
                   int M, int N, int K, int relu) {
    int tiles = (M / 16) * ((N / 16) / 4);
    wmma_gemm_pack<4><<<(tiles + 7) / 8, 256, 0, stream>>>(A, Bp, bias, C, M, N, K, relu);
  };
  auto gemm1 = [&](const float* A, const _Float16* Bp, const float* bias, float* C,
                   int M, int N, int K, int relu) {
    int tiles = (M / 16) * (N / 16);
    wmma_gemm_pack<1><<<(tiles + 7) / 8, 256, 0, stream>>>(A, Bp, bias, C, M, N, K, relu);
  };

  // ---- 3 GIN layers ----
  const float* cur = x;
  for (int l = 0; l < 3; ++l) {
    const int dx = (l == 0) ? inDim : hid;     // true feature width
    const int da = (l == 0) ? 32 : hid;        // padded agg width
    float* agg = (l == 0) ? AGGP : AGG;
    init_agg<<<4096, 256, 0, stream>>>(cur, agg, nNodes, dx, da);        // agg = [x | 0]
    scatter_add_edges<<<8192, 256, 0, stream>>>(cur, srcIdx, dstIdx, agg,
                                                nEdges, dx / 4, dx, da); // agg += sum_j x_j
    gemm4(agg, w1p[l], b1[l], MID, nNodes, hid, da, 1);                  // relu(agg@w1+b1)
    gemm4(MID, w2p[l], b2[l], X, nNodes, hid, hid, 0);                   // @w2+b2
    zero_f32<<<1, 256, 0, stream>>>(SUMS, 2 * hid);
    bn_stats<<<512, 256, 0, stream>>>(X, SUMS, nNodes, hid);
    bn_apply_relu<<<4096, 256, 0, stream>>>(X, SUMS, gm[l], bt[l], nNodes, hid);
    cur = X;
  }

  // ---- global add pool ----
  zero_f32<<<512, 256, 0, stream>>>(GE, G * hid);
  pool_add<<<8192, 256, 0, stream>>>(X, batch, GE, nNodes, hid / 4, hid);

  // ---- readout MLP ----
  gemm4(GE, rw1p, rb1, R1, G, r1d, hid, 1);
  gemm4(R1, rw2p, rb2, R2, G, r2d, r1d, 1);
  gemm1(R2, rw3p, RB3P, R3, G, tasksPad, r2d, 0);
  extract_cols<<<(G * tasks + 255) / 256, 256, 0, stream>>>(R3, (float*)d_out,
                                                            G, tasksPad, tasks);
}
